// CNN_48644799594648
// MI455X (gfx1250) — matrix operand order repaired and checked by reading, not compiled, not run on hardware
//
#include <hip/hip_runtime.h>
#include <hip/hip_bf16.h>
#include <math.h>

// ---------------------------------------------------------------------------
// ConvS2S decoder on MI455X (gfx1250): all GEMMs via v_wmma_f32_16x16x32_bf16
// with double-buffered async global->LDS staging (ASYNCcnt pipeline).
// ---------------------------------------------------------------------------

typedef __attribute__((ext_vector_type(16))) __bf16      v16bf;
typedef __attribute__((ext_vector_type(8)))  float       v8f;
typedef __attribute__((ext_vector_type(4)))  unsigned int u32x4;
typedef unsigned short u16;

#define SQH 0.70710678118654752f

__device__ __forceinline__ u16 f2bf(float x) {
    unsigned int u = __float_as_uint(x);
    u += 0x7FFFu + ((u >> 16) & 1u);          // round-to-nearest-even
    return (u16)(u >> 16);
}

union FragAB { v16bf v; u32x4 q[2]; };
union FragC  { v8f  v; float f[8]; };

// ---- async global->LDS (gfx1250) with synchronous fallback ----------------
#if __has_builtin(__builtin_amdgcn_global_load_async_to_lds_b128)
#define HAVE_ASYNC_LDS 1
typedef __attribute__((__vector_size__(4 * sizeof(int)))) int i32x4v;
typedef __attribute__((address_space(1))) i32x4v* gvp_t;   // global v4i32*
typedef __attribute__((address_space(3))) i32x4v* lvp_t;   // LDS    v4i32*
__device__ __forceinline__ void copy16_lds(u16* lds, const u16* g) {
    __builtin_amdgcn_global_load_async_to_lds_b128((gvp_t)g, (lvp_t)lds, 0, 0);
}
__device__ __forceinline__ void async_wait0() {
#if __has_builtin(__builtin_amdgcn_s_wait_asynccnt)
    __builtin_amdgcn_s_wait_asynccnt(0);
#else
    asm volatile("s_wait_asynccnt 0x0" ::: "memory");
#endif
}
#else
#define HAVE_ASYNC_LDS 0
__device__ __forceinline__ void copy16_lds(u16* lds, const u16* g) {
    *(u32x4*)lds = *(const u32x4*)g;          // global_load_b128 + ds_store_b128
}
__device__ __forceinline__ void async_wait0() {}
#endif

// LDS tile geometry: row stride 40 u16 (= 80B: 64B data + 16B pad, bank-safe)
#define LROW 40
#define LDS_A_ELEMS (128 * LROW)              // 10240 B
#define LDS_B_ELEMS (64 * LROW)               //  5120 B

// C = (A x Bt^T + bias + extra) * scale ; optional bf16 mirror of C.
// Block = 256 thr = 8 waves; block tile 128(M) x 64(N); wave tile 32x32.
// k pipelined in 32-wide slabs, double buffered in LDS via async loads.
__global__ __launch_bounds__(256)
void wmma_gemm_kernel(const u16* __restrict__ A, const u16* __restrict__ Bt,
                      const float* __restrict__ bias,
                      const float* __restrict__ extra,
                      float* __restrict__ C, u16* __restrict__ Cbf,
                      int M, int N, int Kd, float scale,
                      long long sA, long long sB, long long sC, long long sE) {
    __shared__ u16 lA[2][LDS_A_ELEMS];
    __shared__ u16 lB[2][LDS_B_ELEMS];

    const int z = blockIdx.z;
    A += (long long)z * sA;
    Bt += (long long)z * sB;
    C += (long long)z * sC;
    if (extra) extra += (long long)z * sE;
    if (Cbf)   Cbf   += (long long)z * sC;

    const int tid  = threadIdx.x;
    const int lane = tid & 31;
    const int wave = tid >> 5;
    const int mblk = blockIdx.y * 128;
    const int nblk = blockIdx.x * 64;

    // ---- cooperative staging addresses (16B chunks) ----------------------
    // A tile: 128 rows x 4 chunks = 512 chunks -> 2 per thread
    const int arow0 = tid >> 2;               // 0..63
    const int arow1 = arow0 + 64;             // 64..127
    const int sub   = (tid & 3) * 8;          // element offset of 16B chunk
    // B tile: 64 rows x 4 chunks = 256 chunks -> 1 per thread
    const u16* gA0 = A + (size_t)(mblk + arow0) * Kd + sub;
    const u16* gA1 = A + (size_t)(mblk + arow1) * Kd + sub;
    const u16* gB0 = Bt + (size_t)(nblk + arow0) * Kd + sub;
    const int dA0 = arow0 * LROW + sub;
    const int dA1 = arow1 * LROW + sub;
    const int dB0 = arow0 * LROW + sub;

    auto stage = [&](int buf, int k) {
        copy16_lds(&lA[buf][dA0], gA0 + k);
        copy16_lds(&lA[buf][dA1], gA1 + k);
        copy16_lds(&lB[buf][dB0], gB0 + k);
    };

    // ---- per-lane fragment read offsets (u16 elements) -------------------
    const int klo = (lane >> 4) * 8;          // A: 0 or 8 elems (16B)
    const int kb  = (lane >> 4) * 8;          // B: 0 or 8 elems (16B)
    const int aoff0 = ((wave & 3) * 32 + (lane & 15)) * LROW + klo;
    const int aoff1 = aoff0 + 16 * LROW;
    const int boff0 = ((wave >> 2) * 32 + (lane & 15)) * LROW + kb;
    const int boff1 = boff0 + 16 * LROW;

    FragC acc[2][2];
    #pragma unroll
    for (int i = 0; i < 2; ++i)
        #pragma unroll
        for (int j = 0; j < 2; ++j)
            #pragma unroll
            for (int r = 0; r < 8; ++r) acc[i][j].f[r] = 0.0f;

    const int nk = Kd >> 5;                   // k-slabs of 32
    stage(0, 0);                              // prologue: tile 0 in flight

    for (int kk = 0; kk < nk; ++kk) {
        async_wait0();                        // own async loads (tile kk) done
        __syncthreads();                      // tile kk visible; buf^1 free
        const int buf = kk & 1;
        if (kk + 1 < nk) stage(buf ^ 1, (kk + 1) * 32);   // overlap next slab
        if (kk + 2 < nk) {                    // global_prefetch_b8 hints
            __builtin_prefetch(gA0 + (kk + 2) * 32, 0, 0);
            __builtin_prefetch(gB0 + (kk + 2) * 32, 0, 0);
        }

        FragAB a0, a1, b0, b1;                // ds_load_b128 fragment reads
        a0.q[0] = *(const u32x4*)&lA[buf][aoff0];
        a0.q[1] = *(const u32x4*)&lA[buf][aoff0 + 16];
        a1.q[0] = *(const u32x4*)&lA[buf][aoff1];
        a1.q[1] = *(const u32x4*)&lA[buf][aoff1 + 16];
        b0.q[0] = *(const u32x4*)&lB[buf][boff0];
        b0.q[1] = *(const u32x4*)&lB[buf][boff0 + 16];
        b1.q[0] = *(const u32x4*)&lB[buf][boff1];
        b1.q[1] = *(const u32x4*)&lB[buf][boff1 + 16];

        acc[0][0].v = __builtin_amdgcn_wmma_f32_16x16x32_bf16(
            false, a0.v, false, b0.v, (short)0, acc[0][0].v, false, false);
        acc[0][1].v = __builtin_amdgcn_wmma_f32_16x16x32_bf16(
            false, a0.v, false, b1.v, (short)0, acc[0][1].v, false, false);
        acc[1][0].v = __builtin_amdgcn_wmma_f32_16x16x32_bf16(
            false, a1.v, false, b0.v, (short)0, acc[1][0].v, false, false);
        acc[1][1].v = __builtin_amdgcn_wmma_f32_16x16x32_bf16(
            false, a1.v, false, b1.v, (short)0, acc[1][1].v, false, false);
    }

    // C/D layout: VGPR r -> row = r + (lane/16)*8 ; col = lane%16
    const int m0 = mblk + (wave & 3) * 32;
    const int n0 = nblk + (wave >> 2) * 32;
    const int rbase = (lane >> 4) << 3;
    const int coll  = lane & 15;
    #pragma unroll
    for (int fi = 0; fi < 2; ++fi) {
        #pragma unroll
        for (int fj = 0; fj < 2; ++fj) {
            const int col = n0 + fj * 16 + coll;
            const float bc = bias ? bias[col] : 0.0f;
            #pragma unroll
            for (int r = 0; r < 8; ++r) {
                const int row = m0 + fi * 16 + rbase + r;
                float v = acc[fi][fj].f[r] + bc;
                if (extra) v += extra[(size_t)row * N + col];
                v *= scale;
                C[(size_t)row * N + col] = v;
                if (Cbf) Cbf[(size_t)row * N + col] = f2bf(v);
            }
        }
    }
}

// ---------------------------------------------------------------------------
// Support kernels
// ---------------------------------------------------------------------------

__global__ void cvt_bf16_kernel(const float* __restrict__ in,
                                u16* __restrict__ out, int n) {
    int i = blockIdx.x * 256 + threadIdx.x;
    if (i < n) out[i] = f2bf(in[i]);
}

// out_emb[b,t,:] = table[ids[b,t],:] + pe[t,:]
__global__ void embed_kernel(const int* __restrict__ ids,
                             const float* __restrict__ table,
                             const float* __restrict__ pe,
                             float* __restrict__ outf, u16* __restrict__ outb,
                             int T, int H) {
    const int bt = blockIdx.x;
    const int t  = bt & (T - 1);
    const int id = ids[bt];
    const float* src = table + (size_t)id * H;
    const float* pp  = pe + (size_t)t * H;
    float* dst = outf + (size_t)bt * H;
    u16*   db  = outb + (size_t)bt * H;
    for (int h = threadIdx.x; h < H; h += blockDim.x) {
        float v = src[h] + pp[h];
        dst[h] = v;
        db[h] = f2bf(v);
    }
}

// in_emb [B,T,H] f32 -> [B,H,T] bf16 (LDS tiled transpose)
__global__ void transpose_bf_kernel(const float* __restrict__ in,
                                    u16* __restrict__ out, int T, int H) {
    __shared__ float tile[32][33];
    const int b = blockIdx.z;
    const int t0 = blockIdx.x * 32, h0 = blockIdx.y * 32;
    for (int i = threadIdx.y; i < 32; i += 8)
        tile[i][threadIdx.x] = in[((size_t)b * T + t0 + i) * H + h0 + threadIdx.x];
    __syncthreads();
    for (int i = threadIdx.y; i < 32; i += 8)
        out[((size_t)b * H + h0 + i) * T + t0 + threadIdx.x] = f2bf(tile[threadIdx.x][i]);
}

// conv_w[o,c,tap] (OIK) -> bf16 Wt[o, tap*H + c]  (one layer)
__global__ void repack_convw_kernel(const float* __restrict__ w,
                                    u16* __restrict__ out, int H, int Kc) {
    const int i = blockIdx.x * 256 + threadIdx.x;     // over 2H*H (o,c)
    const int o = i / H, c = i - o * H;
    for (int tap = 0; tap < 3; ++tap)
        out[(size_t)o * (3 * H) + tap * H + c] =
            f2bf(w[((size_t)o * H + c) * Kc + tap]);
}

// causal im2col (K=3): Acol[bt, tap*H + c] = h[b, t+tap-2, c] (0 if t+tap<2)
__global__ void im2col_kernel(const u16* __restrict__ hbf,
                              u16* __restrict__ out, int T, int H) {
    const int i = blockIdx.x * 256 + threadIdx.x;     // over B*T*H
    const int bt = i >> 10;                           // H == 1024
    const int c = i & (H - 1);
    const int t = bt & (T - 1);
    const int b = bt / T;
    #pragma unroll
    for (int tap = 0; tap < 3; ++tap) {
        const int ts = t + tap - 2;
        u16 v = (ts >= 0) ? hbf[((size_t)b * T + ts) * H + c] : (u16)0;
        out[(size_t)bt * (3 * H) + tap * H + c] = v;
    }
}

// GLU: act = conv[:, :H] * sigmoid(conv[:, H:2H])
__global__ void glu_kernel(const float* __restrict__ conv,
                           float* __restrict__ act, u16* __restrict__ actbf,
                           int H) {
    const int i = blockIdx.x * 256 + threadIdx.x;     // over M*H
    const int m = i >> 10, c = i & (H - 1);
    const float a = conv[(size_t)m * (2 * H) + c];
    const float g = conv[(size_t)m * (2 * H) + H + c];
    const float v = a / (1.0f + __expf(-g));
    act[i] = v;
    actbf[i] = f2bf(v);
}

// masked softmax over S=512; one block (256 thr) per (b,t) row -> p bf16
__global__ void softmax_kernel(const float* __restrict__ scores,
                               const unsigned char* __restrict__ mask,
                               u16* __restrict__ p) {
    const int row = blockIdx.x;
    const int b = row >> 9;                           // T_OUT == 512
    const int tid = threadIdx.x;
    const float* s = scores + (size_t)row * 512;
    const unsigned char* mk = mask + (size_t)b * 512;
    __shared__ float red[8];

    float v0 = mk[tid]       ? -__builtin_inff() : s[tid];
    float v1 = mk[tid + 256] ? -__builtin_inff() : s[tid + 256];

    float m = fmaxf(v0, v1);
    #pragma unroll
    for (int o = 16; o > 0; o >>= 1) m = fmaxf(m, __shfl_down(m, o, 32));
    if ((tid & 31) == 0) red[tid >> 5] = m;
    __syncthreads();
    if (tid == 0) {
        float mm = red[0];
        #pragma unroll
        for (int i = 1; i < 8; ++i) mm = fmaxf(mm, red[i]);
        red[0] = mm;
    }
    __syncthreads();
    m = red[0];
    __syncthreads();

    const float e0 = __expf(v0 - m);
    const float e1 = __expf(v1 - m);
    float sum = e0 + e1;
    #pragma unroll
    for (int o = 16; o > 0; o >>= 1) sum += __shfl_down(sum, o, 32);
    if ((tid & 31) == 0) red[tid >> 5] = sum;
    __syncthreads();
    if (tid == 0) {
        float ss = red[0];
        #pragma unroll
        for (int i = 1; i < 8; ++i) ss += red[i];
        red[0] = ss;
    }
    __syncthreads();
    const float inv = 1.0f / red[0];
    p[(size_t)row * 512 + tid]       = f2bf(e0 * inv);
    p[(size_t)row * 512 + tid + 256] = f2bf(e1 * inv);
}

// h = ((attn + act)*SQH + resid)*SQH ; write f32 + bf16
__global__ void combine_kernel(const float* __restrict__ attn,
                               const float* __restrict__ act,
                               const float* __restrict__ resid,
                               float* __restrict__ h, u16* __restrict__ hbf) {
    const int i = blockIdx.x * 256 + threadIdx.x;
    const float v = ((attn[i] + act[i]) * SQH + resid[i]) * SQH;
    h[i] = v;
    hbf[i] = f2bf(v);
}

// ---------------------------------------------------------------------------
// Host orchestration
// ---------------------------------------------------------------------------

extern "C" void kernel_launch(void* const* d_in, const int* in_sizes, int n_in,
                              void* d_out, int out_size, void* d_ws, size_t ws_size,
                              hipStream_t stream) {
    (void)in_sizes; (void)n_in; (void)out_size; (void)ws_size;

    constexpr int Bz = 8, T = 512, H = 1024, Kc = 3, Ln = 6;
    constexpr int M = Bz * T;                  // 4096

    const int*           ids    = (const int*)d_in[0];
    const float*         enc    = (const float*)d_in[1];
    const float*         inemb  = (const float*)d_in[2];
    const unsigned char* mask   = (const unsigned char*)d_in[3];
    const float*         table  = (const float*)d_in[4];
    const float*         pe     = (const float*)d_in[5];
    const float*         W_in   = (const float*)d_in[6];
    const float*         b_in   = (const float*)d_in[7];
    const float*         W_out  = (const float*)d_in[8];
    const float*         b_out  = (const float*)d_in[9];
    const float*         W_proj = (const float*)d_in[10];
    const float*         b_proj = (const float*)d_in[11];
    const float*         conv_w = (const float*)d_in[12];
    const float*         conv_b = (const float*)d_in[13];
    const float*         W_att  = (const float*)d_in[14];
    const float*         b_att  = (const float*)d_in[15];
    float*               out    = (float*)d_out;

    // ---- workspace carve-out (256B aligned) ------------------------------
    char* base = (char*)d_ws;
    size_t off = 0;
    auto alloc = [&](size_t bytes) -> char* {
        char* p = base + off;
        off = (off + bytes + 255) & ~(size_t)255;
        return p;
    };
    float* out_emb = (float*)alloc((size_t)M * H * 4);
    float* hbuf    = (float*)alloc((size_t)M * H * 4);
    float* resid   = (float*)alloc((size_t)M * H * 4);
    float* convout = (float*)alloc((size_t)M * 2 * H * 4);  // reused below
    float* act     = (float*)alloc((size_t)M * H * 4);
    float* scores  = (float*)alloc((size_t)Bz * T * T * 4);
    u16* out_embB  = (u16*)alloc((size_t)M * H * 2);
    u16* hB        = (u16*)alloc((size_t)M * H * 2);
    u16* actB      = (u16*)alloc((size_t)M * H * 2);
    u16* dB        = (u16*)alloc((size_t)M * H * 2);
    u16* pB        = (u16*)alloc((size_t)Bz * T * T * 2);
    u16* encB      = (u16*)alloc((size_t)Bz * T * H * 2);
    u16* inembT    = (u16*)alloc((size_t)Bz * H * T * 2);
    u16* im2colB   = (u16*)alloc((size_t)M * 3 * H * 2);
    u16* WinB      = (u16*)alloc((size_t)H * H * 2);
    u16* WoutB     = (u16*)alloc((size_t)H * H * 2);
    u16* WlayB     = (u16*)alloc((size_t)H * H * 2);
    u16* convwB    = (u16*)alloc((size_t)2 * H * 3 * H * 2);
    // convout region re-used after GLU consumes it:
    float* dtmp    = convout;                       // [M,H]
    float* attnraw = convout + (size_t)M * H;       // [M,H]

    auto gemm = [&](const u16* A, const u16* Bt, const float* bias,
                    const float* extra, float* C, u16* Cbf,
                    int Mm, int Nn, int Kd, float scale,
                    long long sA, long long sB, long long sC, long long sE,
                    int nb) {
        dim3 g(Nn / 64, Mm / 128, nb);
        wmma_gemm_kernel<<<g, 256, 0, stream>>>(A, Bt, bias, extra, C, Cbf,
                                                Mm, Nn, Kd, scale, sA, sB, sC, sE);
    };
    auto cvt = [&](const float* in, u16* o, int n) {
        cvt_bf16_kernel<<<(n + 255) / 256, 256, 0, stream>>>(in, o, n);
    };

    // ---- prep ------------------------------------------------------------
    embed_kernel<<<M, 256, 0, stream>>>(ids, table, pe, out_emb, out_embB, T, H);
    cvt(W_in, WinB, H * H);
    cvt(W_out, WoutB, H * H);
    cvt(enc, encB, Bz * T * H);
    transpose_bf_kernel<<<dim3(T / 32, H / 32, Bz), dim3(32, 8), 0, stream>>>(
        inemb, inembT, T, H);

    // h = out_emb @ W_in^T + b_in
    gemm(out_embB, WinB, b_in, nullptr, hbuf, hB, M, H, H, 1.0f, 0, 0, 0, 0, 1);

    const float attn_scale = sqrtf((float)T);       // T_in * sqrt(1/T_in)

    for (int l = 0; l < Ln; ++l) {
        // residual projection (layers 1..L-1)
        if (l > 0) {
            cvt(W_proj + (size_t)(l - 1) * H * H, WlayB, H * H);
            gemm(hB, WlayB, b_proj + (size_t)(l - 1) * H, nullptr,
                 resid, nullptr, M, H, H, 1.0f, 0, 0, 0, 0, 1);
        }
        // causal conv as GEMM: [M, 3H] x [2H, 3H]^T
        repack_convw_kernel<<<(2 * H * H) / 256, 256, 0, stream>>>(
            conv_w + (size_t)l * 2 * H * H * Kc, convwB, H, Kc);
        im2col_kernel<<<(M * H) / 256, 256, 0, stream>>>(hB, im2colB, T, H);
        gemm(im2colB, convwB, conv_b + (size_t)l * 2 * H, nullptr,
             convout, nullptr, M, 2 * H, 3 * H, 1.0f, 0, 0, 0, 0, 1);
        glu_kernel<<<(M * H) / 256, 256, 0, stream>>>(convout, act, actB, H);

        // attention: d = (act @ Wa^T + ba + out_emb) * sqrt(0.5)
        cvt(W_att + (size_t)l * H * H, WlayB, H * H);
        gemm(actB, WlayB, b_att + (size_t)l * H, out_emb,
             dtmp, dB, M, H, H, SQH, 0, 0, 0, 0, 1);

        // scores[b] = d[b] @ enc[b]^T    (batched, 8 x [512,512,1024])
        gemm(dB, encB, nullptr, nullptr, scores, nullptr,
             T, T, H, 1.0f, (long long)T * H, (long long)T * H,
             (long long)T * T, 0, Bz);
        softmax_kernel<<<Bz * T, 256, 0, stream>>>(scores, mask, pB);

        // attn[b] = p[b] @ in_emb[b] * sqrt(T)   (batched, 8 x [512,1024,512])
        gemm(pB, inembT, nullptr, nullptr, attnraw, nullptr,
             T, H, T, attn_scale, (long long)T * T, (long long)H * T,
             (long long)T * H, 0, Bz);

        // h = ((attn + act)*SQH + resid)*SQH
        combine_kernel<<<(M * H) / 256, 256, 0, stream>>>(
            attnraw, act, (l == 0) ? hbuf : resid, hbuf, hB);
    }

    // out = h @ W_out^T + b_out
    gemm(hB, WoutB, b_out, nullptr, out, nullptr, M, H, H, 1.0f, 0, 0, 0, 0, 1);
}
